// MultiHeadAttention_27023934226588
// MI455X (gfx1250) — matrix operand order repaired and checked by reading, not compiled, not run on hardware
//
#include <hip/hip_runtime.h>

// ---------------------------------------------------------------------------
// MHA forward for MI455X (gfx1250), bf16 WMMA + flash attention,
// register-blocked for A/B fragment reuse (compute-bound regime).
// ---------------------------------------------------------------------------

typedef __attribute__((ext_vector_type(16))) __bf16 v16bf;
typedef __attribute__((ext_vector_type(8)))  __bf16 v8bf;
typedef __attribute__((ext_vector_type(8)))  float  v8f;

#define WMMA_BF16(a, b, c) \
  __builtin_amdgcn_wmma_f32_16x16x32_bf16(false, (a), false, (b), (short)0, (c), false, false)

namespace cfg {
constexpr int DM   = 768;      // d_model
constexpr int NH   = 12;       // heads
constexpr int DK   = 64;       // head dim
constexpr int Bn   = 4;        // batch
constexpr int Sn   = 2048;     // seq
constexpr int M    = Bn * Sn;  // 8192 rows
constexpr int QKVN = 3 * DM;   // 2304
}

// ---- WMMA fragment loaders (layouts per CDNA5 ISA §7.12.2) ------
// A (16x32 bf16): lane&15 = row; lanes<16 hold K 0..7 & 16..23, lanes>=16 hold
// K 8..15 & 24..31.  base = row-major tile start, ld = row stride (elements).
__device__ inline v16bf load_a16(const __bf16* base, int ld, int lane) {
  const int r    = lane & 15;
  const int koff = (lane >> 4) << 3;   // 0 or 8
  const __bf16* p = base + (size_t)r * ld;
  union { v16bf v; v8bf h[2]; } u;
  u.h[0] = *(const v8bf*)(p + koff);        // K = koff .. koff+7
  u.h[1] = *(const v8bf*)(p + 16 + koff);   // K = 16+koff .. 16+koff+7
  return u.v;
}

// B (32x16 bf16): lane&15 = column n; lanes<16 hold K 0..7 & 16..23, lanes>=16
// hold K 8..15 & 24..31.  baseT points at B^T stored row-major [n][k].
__device__ inline v16bf load_b16(const __bf16* baseT, int ld, int lane) {
  const int n  = lane & 15;
  const int kh = (lane >> 4) << 3;     // 0 or 8
  const __bf16* p = baseT + (size_t)n * ld;
  union { v16bf v; v8bf h[2]; } u;
  u.h[0] = *(const v8bf*)(p + kh);
  u.h[1] = *(const v8bf*)(p + 16 + kh);
  return u.v;
}

// ---- cross-lane row reductions over the 16-lane halves (C layout rows) -----
__device__ inline float redmax16(float v) {
#pragma unroll
  for (int m = 1; m < 16; m <<= 1) v = fmaxf(v, __shfl_xor(v, m, 32));
  return v;
}
__device__ inline float redsum16(float v) {
#pragma unroll
  for (int m = 1; m < 16; m <<= 1) v += __shfl_xor(v, m, 32);
  return v;
}

// ---------------------------------------------------------------------------
// Prep kernels: f32 -> bf16 convert / transpose-convert.
// ---------------------------------------------------------------------------
__global__ void convert_bf16(const float* __restrict__ src, __bf16* __restrict__ dst, int n) {
  int i = blockIdx.x * blockDim.x + threadIdx.x;
  if (i < n) dst[i] = (__bf16)src[i];
}

// dst[c][r] = src[r][c] ; src is rows x cols row-major.
__global__ void transpose_bf16(const float* __restrict__ src, __bf16* __restrict__ dst,
                               int rows, int cols) {
  int i = blockIdx.x * blockDim.x + threadIdx.x;
  if (i >= rows * cols) return;
  int r = i / cols, c = i - r * cols;
  dst[(size_t)c * rows + r] = (__bf16)src[i];
}

// ---------------------------------------------------------------------------
// QKV projection: [8192x768] @ [768x2304] + bias.  One wave = 64x64 output
// tile (4x4 WMMA sub-tiles, 16 WMMAs per K=32 step, 1 b128 load per WMMA).
// Epilogue scatters to Q/K ([b,h,s,d]) and V^T ([b,h,d,s]) in bf16.
// ---------------------------------------------------------------------------
__global__ __launch_bounds__(256) void qkv_gemm(const __bf16* __restrict__ Xb,
                                                const __bf16* __restrict__ WT,
                                                const float*  __restrict__ bias,
                                                __bf16* __restrict__ Qb,
                                                __bf16* __restrict__ Kb,
                                                __bf16* __restrict__ Vt) {
  using namespace cfg;
  const int lane = threadIdx.x & 31;
  const int wid  = threadIdx.x >> 5;
  const int tile = blockIdx.x * 8 + wid;
  const int NT   = QKVN / 64;            // 36 col super-tiles
  const int tn = tile % NT, tm = tile / NT;
  const int m0 = tm * 64, n0 = tn * 64;

  v8f acc[4][4] = {};
#pragma unroll 1
  for (int kk = 0; kk < DM; kk += 32) {
    v16bf a[4], b[4];
#pragma unroll
    for (int i = 0; i < 4; ++i)
      a[i] = load_a16(Xb + (size_t)(m0 + 16 * i) * DM + kk, DM, lane);
#pragma unroll
    for (int j = 0; j < 4; ++j)
      b[j] = load_b16(WT + (size_t)(n0 + 16 * j) * DM + kk, DM, lane);
#pragma unroll
    for (int i = 0; i < 4; ++i)
#pragma unroll
      for (int j = 0; j < 4; ++j)
        acc[i][j] = WMMA_BF16(a[i], b[j], acc[i][j]);
  }

  const int half = lane >> 4, n = lane & 15;
#pragma unroll
  for (int jn = 0; jn < 4; ++jn) {
    const int col  = n0 + 16 * jn + n;
    const int part = col / DM;           // 0=q 1=k 2=v (uniform per sub-tile)
    const int rem  = col - part * DM;
    const int h    = rem >> 6;           // uniform per sub-tile
    const int d    = rem & 63;
    const float bv = bias[col];
#pragma unroll
    for (int im = 0; im < 4; ++im) {
#pragma unroll
      for (int j = 0; j < 8; ++j) {
        const int row = m0 + 16 * im + j + 8 * half;
        const int b_  = row >> 11;       // / 2048
        const int s_  = row & 2047;
        const __bf16 v = (__bf16)(acc[im][jn][j] + bv);
        const size_t head = (size_t)(b_ * NH + h);
        if (part == 0)      Qb[(head * Sn + s_) * DK + d] = v;
        else if (part == 1) Kb[(head * Sn + s_) * DK + d] = v;
        else                Vt[(head * DK + d) * Sn + s_] = v;
      }
    }
  }
}

// ---------------------------------------------------------------------------
// Flash attention: one wave per (b, h, 32-query tile); causal; 32 keys/step.
// K/V B-fragments are shared across the two 16-row query sub-tiles
// (16 WMMAs per 16 K/V b128 loads).  8 waves/block, private LDS slices.
// ---------------------------------------------------------------------------
__global__ __launch_bounds__(256) void attn_fa(const __bf16* __restrict__ Qb,
                                               const __bf16* __restrict__ Kb,
                                               const __bf16* __restrict__ Vt,
                                               __bf16* __restrict__ Ao) {
  using namespace cfg;
  __shared__ __align__(64) __bf16 ldsAll[8 * 32 * 32];
  const int lane = threadIdx.x & 31;
  const int wid  = threadIdx.x >> 5;
  __bf16* lds = &ldsAll[wid * 32 * 32];

  const int tile   = blockIdx.x * 8 + wid;
  const int QTILES = Sn / 32;                   // 64 query super-tiles
  const int qt = tile % QTILES;
  const int h  = (tile / QTILES) % NH;
  const int b  = tile / (QTILES * NH);
  const int q0 = qt * 32;

  const __bf16* Qh = Qb + (size_t)(b * NH + h) * Sn * DK;
  const __bf16* Kh = Kb + (size_t)(b * NH + h) * Sn * DK;
  const __bf16* Vh = Vt + (size_t)(b * NH + h) * DK * Sn;

  // Q fragments: [sub-tile t][d chunk 0..31 / 32..63]
  v16bf aQ[2][2];
#pragma unroll
  for (int t = 0; t < 2; ++t) {
    aQ[t][0] = load_a16(Qh + (size_t)(q0 + 16 * t) * DK + 0,  DK, lane);
    aQ[t][1] = load_a16(Qh + (size_t)(q0 + 16 * t) * DK + 32, DK, lane);
  }

  v8f acc[2][4] = {};
  float mrow[2][8], lrow[2][8];
#pragma unroll
  for (int t = 0; t < 2; ++t)
#pragma unroll
    for (int j = 0; j < 8; ++j) { mrow[t][j] = -1e30f; lrow[t][j] = 0.f; }

  const int half = lane >> 4, n = lane & 15;
  const int kend = q0 + 32;                      // causal bound (exclusive)

#pragma unroll 1
  for (int kb = 0; kb < kend; kb += 32) {
    // Shared B-fragments for this key block.
    const v16bf bK00 = load_b16(Kh + (size_t)kb * DK + 0,         DK, lane);
    const v16bf bK01 = load_b16(Kh + (size_t)kb * DK + 32,        DK, lane);
    const v16bf bK10 = load_b16(Kh + (size_t)(kb + 16) * DK + 0,  DK, lane);
    const v16bf bK11 = load_b16(Kh + (size_t)(kb + 16) * DK + 32, DK, lane);
    const v16bf bV0  = load_b16(Vh + (size_t)0  * Sn + kb, Sn, lane);
    const v16bf bV1  = load_b16(Vh + (size_t)16 * Sn + kb, Sn, lane);
    const v16bf bV2  = load_b16(Vh + (size_t)32 * Sn + kb, Sn, lane);
    const v16bf bV3  = load_b16(Vh + (size_t)48 * Sn + kb, Sn, lane);

#pragma unroll
    for (int t = 0; t < 2; ++t) {
      // ---- scores: two 16x16 C tiles covering keys kb..kb+31 ----
      v8f c0 = {}, c1 = {};
      c0 = WMMA_BF16(aQ[t][0], bK00, c0);
      c0 = WMMA_BF16(aQ[t][1], bK01, c0);
      c1 = WMMA_BF16(aQ[t][0], bK10, c1);
      c1 = WMMA_BF16(aQ[t][1], bK11, c1);

      float p0[8], p1[8];
#pragma unroll
      for (int j = 0; j < 8; ++j) {
        const int row = q0 + 16 * t + j + 8 * half;
        const int k0i = kb + n, k1i = kb + 16 + n;
        float s0 = c0[j] * 0.125f;               // 1/sqrt(64)
        float s1 = c1[j] * 0.125f;
        if (k0i > row) s0 = -1e30f;              // causal mask
        if (k1i > row) s1 = -1e30f;
        float mnew = fmaxf(mrow[t][j], redmax16(fmaxf(s0, s1)));
        float corr = __expf(mrow[t][j] - mnew);
        mrow[t][j] = mnew;
        float e0 = __expf(s0 - mnew), e1 = __expf(s1 - mnew);
        lrow[t][j] = lrow[t][j] * corr + redsum16(e0 + e1);
        acc[t][0][j] *= corr; acc[t][1][j] *= corr;
        acc[t][2][j] *= corr; acc[t][3][j] *= corr;
        p0[j] = e0; p1[j] = e1;
      }

      // ---- C-layout -> A-layout transpose of P via private LDS slice ----
      __bf16* ldsT = lds + t * (16 * 32);
#pragma unroll
      for (int j = 0; j < 8; ++j) {
        const int r = j + 8 * half;
        ldsT[r * 32 + n]      = (__bf16)p0[j];
        ldsT[r * 32 + 16 + n] = (__bf16)p1[j];
      }
      asm volatile("s_wait_dscnt 0" ::: "memory"); // same-wave LDS RAW
      const v16bf aP = load_a16(ldsT, 32, lane);

      // ---- P(16x32) @ V(32x64): four 16-column d tiles ----
      acc[t][0] = WMMA_BF16(aP, bV0, acc[t][0]);
      acc[t][1] = WMMA_BF16(aP, bV1, acc[t][1]);
      acc[t][2] = WMMA_BF16(aP, bV2, acc[t][2]);
      acc[t][3] = WMMA_BF16(aP, bV3, acc[t][3]);
    }
  }

  // ---- epilogue: normalize and store bf16 to Ao[b, s, h*64 + d] ----
#pragma unroll
  for (int t = 0; t < 2; ++t) {
#pragma unroll
    for (int j = 0; j < 8; ++j) {
      const int row = q0 + 16 * t + j + 8 * half;
      const float inv = 1.0f / lrow[t][j];
      __bf16* dst = Ao + ((size_t)b * Sn + row) * DM + h * DK + n;
      dst[0]  = (__bf16)(acc[t][0][j] * inv);
      dst[16] = (__bf16)(acc[t][1][j] * inv);
      dst[32] = (__bf16)(acc[t][2][j] * inv);
      dst[48] = (__bf16)(acc[t][3][j] * inv);
    }
  }
}

// ---------------------------------------------------------------------------
// Output projection: [8192x768] @ [768x768] + bias -> f32 d_out.
// One wave = 64x32 output tile (4x2 sub-tiles, 8 WMMAs per K=32 step).
// ---------------------------------------------------------------------------
__global__ __launch_bounds__(256) void out_gemm(const __bf16* __restrict__ Ab,
                                                const __bf16* __restrict__ WT,
                                                const float*  __restrict__ bias,
                                                float* __restrict__ out) {
  using namespace cfg;
  const int lane = threadIdx.x & 31;
  const int wid  = threadIdx.x >> 5;
  const int tile = blockIdx.x * 8 + wid;
  const int NT   = DM / 32;                 // 24 col super-tiles
  const int tn = tile % NT, tm = tile / NT;
  const int m0 = tm * 64, n0 = tn * 32;

  v8f acc[4][2] = {};
#pragma unroll 1
  for (int kk = 0; kk < DM; kk += 32) {
    v16bf a[4], b[2];
#pragma unroll
    for (int i = 0; i < 4; ++i)
      a[i] = load_a16(Ab + (size_t)(m0 + 16 * i) * DM + kk, DM, lane);
#pragma unroll
    for (int j = 0; j < 2; ++j)
      b[j] = load_b16(WT + (size_t)(n0 + 16 * j) * DM + kk, DM, lane);
#pragma unroll
    for (int i = 0; i < 4; ++i)
#pragma unroll
      for (int j = 0; j < 2; ++j)
        acc[i][j] = WMMA_BF16(a[i], b[j], acc[i][j]);
  }

  const int half = lane >> 4, n = lane & 15;
#pragma unroll
  for (int jn = 0; jn < 2; ++jn) {
    const float bv = bias[n0 + 16 * jn + n];
#pragma unroll
    for (int im = 0; im < 4; ++im) {
#pragma unroll
      for (int j = 0; j < 8; ++j) {
        const int row = m0 + 16 * im + j + 8 * half;
        out[(size_t)row * DM + n0 + 16 * jn + n] = acc[im][jn][j] + bv;
      }
    }
  }
}

// ---------------------------------------------------------------------------
extern "C" void kernel_launch(void* const* d_in, const int* in_sizes, int n_in,
                              void* d_out, int out_size, void* d_ws, size_t ws_size,
                              hipStream_t stream) {
  using namespace cfg;
  const float* x    = (const float*)d_in[0];
  // d_in[1] = causal mask (applied analytically in attn_fa)
  const float* Wqkv = (const float*)d_in[2];
  const float* bqkv = (const float*)d_in[3];
  const float* Wout = (const float*)d_in[4];
  const float* bout = (const float*)d_in[5];
  float* out = (float*)d_out;

  char* ws = (char*)d_ws;
  size_t off = 0;
  auto take = [&](size_t elems) -> __bf16* {
    __bf16* p = (__bf16*)(ws + off);
    off += ((elems * sizeof(__bf16) + 255) / 256) * 256;
    return p;
  };
  __bf16* Xb    = take((size_t)M * DM);        // x in bf16
  __bf16* WqkvT = take((size_t)QKVN * DM);     // Wqkv^T bf16
  __bf16* WoutT = take((size_t)DM * DM);       // Wout^T bf16
  __bf16* Qb    = take((size_t)Bn * NH * Sn * DK);
  __bf16* Kb    = take((size_t)Bn * NH * Sn * DK);
  __bf16* Vt    = take((size_t)Bn * NH * DK * Sn);
  __bf16* Ao    = take((size_t)M * DM);        // attention output bf16

  const int nX = M * DM;                        // 6,291,456
  convert_bf16<<<(nX + 255) / 256, 256, 0, stream>>>(x, Xb, nX);
  transpose_bf16<<<(DM * QKVN + 255) / 256, 256, 0, stream>>>(Wqkv, WqkvT, DM, QKVN);
  transpose_bf16<<<(DM * DM + 255) / 256, 256, 0, stream>>>(Wout, WoutT, DM, DM);

  // (8192/64)*(2304/64) = 4608 wave-tiles / 8 waves = 576 blocks
  qkv_gemm<<<576, 256, 0, stream>>>(Xb, WqkvT, bqkv, Qb, Kb, Vt);

  // B*H*(S/32) = 3072 wave-tiles / 8 waves = 384 blocks
  attn_fa<<<384, 256, 0, stream>>>(Qb, Kb, Vt, Ao);

  // (8192/64)*(768/32) = 3072 wave-tiles / 8 waves = 384 blocks
  out_gemm<<<384, 256, 0, stream>>>(Ao, WoutT, bout, out);
}
